// MHA_80307298500800
// MI455X (gfx1250) — matrix operand order repaired and checked by reading, not compiled, not run on hardware
//
#include <hip/hip_runtime.h>
#include <hip/hip_bf16.h>

// MHA-like pipeline for MI455X (gfx1250, wave32, WMMA).
// Heavy math on v_wmma_f32_16x16x32_bf16 (bf16 in / f32 accumulate).
// GEMM staging: CDNA5 per-lane async DMA (global_load_async_to_lds_b128,
// ASYNCcnt), double-buffered. Attention K staging: Tensor Data Mover
// (tensor_load_to_lds + s_wait_tensorcnt) with hardware LDS row padding.

typedef __attribute__((ext_vector_type(16))) __bf16 v16bf;
typedef __attribute__((ext_vector_type(8)))  __bf16 v8bf;
typedef __attribute__((ext_vector_type(8)))  float   v8f;
typedef __attribute__((ext_vector_type(4)))  unsigned v4u;
typedef __attribute__((ext_vector_type(8)))  unsigned v8u;

#define B_  2
#define S_  2048
#define D_  2048
#define H_  16
#define DH_ 128
#define M_  (B_*S_)   // 4096 rows

__device__ __forceinline__ __bf16 f2bf(float f) {
  unsigned u = __builtin_bit_cast(unsigned, f);
  unsigned r = (u + 0x7FFFu + ((u >> 16) & 1u)) >> 16;
  unsigned short s = (unsigned short)r;
  return __builtin_bit_cast(__bf16, s);
}

__device__ __forceinline__ v8f wmma_bf16(v16bf a, v16bf b, v8f c) {
  // (neg_a, A, neg_b, B, c_mod, C, reuse_a, reuse_b)
  return __builtin_amdgcn_wmma_f32_16x16x32_bf16(false, a, false, b, (short)0, c,
                                                 false, false);
}

// CDNA5 async DMA: global -> LDS, 16 bytes per lane, tracked by ASYNCcnt.
__device__ __forceinline__ void async_g2l_b128(unsigned lds_addr, const void* g) {
  asm volatile("global_load_async_to_lds_b128 %0, %1, off"
               :: "v"(lds_addr), "v"((unsigned long long)(size_t)g)
               : "memory");
}
__device__ __forceinline__ void async_wait0() {
  asm volatile("s_wait_asynccnt 0x0" ::: "memory");
}

// CDNA5 Tensor Data Mover: one wave-level 2D tile DMA, global -> LDS.
// Stages a 64-row x 256-byte tile (contiguous rows) and pads LDS by 16 bytes
// after every 256-byte row (LDS row stride 272 B = 136 halves).
__device__ __forceinline__ void tdm_stage_k64x128(const __bf16* gptr, unsigned lds) {
  unsigned long long ga = (unsigned long long)(size_t)gptr;
  v4u g0 = { 1u,                                   // count=1, user descriptor
             lds,                                  // lds_addr (bytes)
             (unsigned)ga,                         // global_addr[31:0]
             ((unsigned)(ga >> 32) & 0x1FFFFFFu)   // global_addr[56:32]
               | (2u << 30) };                     // type=2 ("image")
  v8u g1 = { 0x07520000u,  // data_size=4B | pad_en | pad_interval=5 | pad_amount=3
             32u << 16,    // tensor_dim0 = 32 dwords       (bits 63:48)
             64u << 16,    // tensor_dim1 = 64 rows         (bits 95:80)
             32u << 16,    // tile_dim0   = 32 dwords       (bits 127:112)
             64u,          // tile_dim1   = 64, tile_dim2=0 (bits 143:128)
             32u,          // tensor_dim0_stride = 32 dwords (bits 191:160)
             0u, 0u };
  asm volatile("tensor_load_to_lds %0, %1" :: "s"(g0), "s"(g1) : "memory");
}
__device__ __forceinline__ void tensor_wait0() {
  __builtin_amdgcn_s_wait_tensorcnt(0);
}

// ---------------- fp32 -> bf16 conversion ----------------
__global__ __launch_bounds__(256) void k_cvt(const float* __restrict__ s,
                                             __bf16* __restrict__ d, int n) {
  int i = (blockIdx.x * blockDim.x + threadIdx.x) * 4;
  if (i + 3 < n) {
    float4 f = *(const float4*)(s + i);
    d[i]     = f2bf(f.x);
    d[i + 1] = f2bf(f.y);
    d[i + 2] = f2bf(f.z);
    d[i + 3] = f2bf(f.w);
  }
}

// ---------------- C = A (MxK) * B^T (NxK rows), fp32 out ----------------
// 128x128 block tile, 8 waves, each wave 32x64 (2x4 WMMA tiles), K-step 32,
// double-buffered async LDS staging.
__global__ __launch_bounds__(256) void k_gemm_bt(const __bf16* __restrict__ A,
                                                 const __bf16* __restrict__ Bm,
                                                 float* __restrict__ C,
                                                 int M, int N, int K) {
  __shared__ alignas(16) __bf16 As[2][128 * 40];  // stride 40: conflict-free b128
  __shared__ alignas(16) __bf16 Bs[2][128 * 40];
  const int t = threadIdx.x, lane = t & 31, wid = t >> 5;
  const int lm = lane & 15, lk = lane >> 4;       // M/N index, K-half
  const int wm = wid & 3, wn = wid >> 2;          // wave: 4 M-bands x 2 N-bands
  const int bm = blockIdx.y * 128, bn = blockIdx.x * 128;

  v8f acc[2][4];
  for (int i = 0; i < 2; ++i)
    for (int j = 0; j < 4; ++j)
      acc[i][j] = (v8f){0.f, 0.f, 0.f, 0.f, 0.f, 0.f, 0.f, 0.f};

  auto stage = [&](int buf, int k0) {
    for (int c = t; c < 512; c += 256) {   // 128 rows x 4 chunks of 8 halves
      int r = c >> 2, cc = (c & 3) * 8;
      async_g2l_b128((unsigned)(size_t)&As[buf][r * 40 + cc],
                     A + (size_t)(bm + r) * K + k0 + cc);
      async_g2l_b128((unsigned)(size_t)&Bs[buf][r * 40 + cc],
                     Bm + (size_t)(bn + r) * K + k0 + cc);
    }
  };

  stage(0, 0);
  async_wait0();
  __syncthreads();

  int cur = 0;
  for (int k0 = 0; k0 < K; k0 += 32) {
    if (k0 + 32 < K) stage(cur ^ 1, k0 + 32);   // prefetch next tile (async)

    v16bf af[2], bfr[4];
    for (int i = 0; i < 2; ++i) {
      int row = wm * 32 + i * 16 + lm;
      union { v16bf v; v8bf h2[2]; } u;
      u.h2[0] = *(const v8bf*)&As[cur][row * 40 + lk * 8];       // K=lk*8+0..7
      u.h2[1] = *(const v8bf*)&As[cur][row * 40 + lk * 8 + 16];  // K=lk*8+16..23
      af[i] = u.v;
    }
    for (int j = 0; j < 4; ++j) {
      int row = wn * 64 + j * 16 + lm;
      union { v16bf v; v8bf h2[2]; } u;
      u.h2[0] = *(const v8bf*)&Bs[cur][row * 40 + lk * 8];       // K=lk*8+0..7
      u.h2[1] = *(const v8bf*)&Bs[cur][row * 40 + lk * 8 + 16];  // K=lk*8+16..23
      bfr[j] = u.v;
    }
    for (int i = 0; i < 2; ++i)
      for (int j = 0; j < 4; ++j)
        acc[i][j] = wmma_bf16(af[i], bfr[j], acc[i][j]);

    async_wait0();      // my prefetch DMA landed
    __syncthreads();    // lowers with s_wait_dscnt 0 before signal: reads done
    cur ^= 1;
  }

  // C/D layout: VGPR e -> M = e + 8*(lane>>4), N = lane&15.
  for (int i = 0; i < 2; ++i)
    for (int j = 0; j < 4; ++j) {
      int rbase = bm + wm * 32 + i * 16 + 8 * lk;
      int col = bn + wn * 64 + j * 16 + lm;
      for (int e = 0; e < 8; ++e)
        C[(size_t)(rbase + e) * N + col] = acc[i][j][e];
    }
}

// ---------------- RoPE + pack Q,K to bf16 head-major (B,H,S,DH) ----------------
__global__ __launch_bounds__(256) void k_rope(const float* __restrict__ Qf,
                                              const float* __restrict__ Kf,
                                              const float* __restrict__ cosT,
                                              const float* __restrict__ sinT,
                                              __bf16* __restrict__ Qb,
                                              __bf16* __restrict__ Kb) {
  int idx = blockIdx.x * blockDim.x + threadIdx.x;  // one (re,im) pair
  int i = idx & 63;
  int h = (idx >> 6) & 15;
  int s = (idx >> 10) & 2047;
  int b = idx >> 21;
  float c = cosT[s * 64 + i], sn = sinT[s * 64 + i];
  size_t src = ((size_t)(b * S_ + s)) * D_ + h * DH_ + 2 * i;
  float qe = Qf[src], qo = Qf[src + 1];
  float ke = Kf[src], ko = Kf[src + 1];
  size_t dst = (((size_t)(b * H_ + h)) * S_ + s) * DH_ + 2 * i;
  Qb[dst]     = f2bf(qe * c - qo * sn);
  Qb[dst + 1] = f2bf(qe * sn + qo * c);
  Kb[dst]     = f2bf(ke * c - ko * sn);
  Kb[dst + 1] = f2bf(ke * sn + ko * c);
}

// ---------------- per-head LSE + diag + w*v ----------------
// grid (S/128, B*H), 8 waves; each wave owns 16 q-rows. K staged in LDS in
// 64-row blocks via the Tensor Data Mover, double-buffered (wave 0 issues).
__global__ __launch_bounds__(256) void k_attn(const __bf16* __restrict__ Qb,
                                              const __bf16* __restrict__ Kb,
                                              const float* __restrict__ Vf,
                                              __bf16* __restrict__ attnb) {
  __shared__ alignas(16) __bf16 Ks[2][64 * 136];  // stride-136 (TDM-padded rows)
  __shared__ float wsm[8 * 16];
  const int t = threadIdx.x, lane = t & 31, wid = t >> 5;
  const int lm = lane & 15, lk = lane >> 4;
  const int bh = blockIdx.y;
  const int b = bh >> 4, h = bh & 15;
  const int q0 = blockIdx.x * 128 + wid * 16;
  const float scale = 0.0883883476483184405f;  // 1/sqrt(128)
  const __bf16* kbase = Kb + (size_t)bh * S_ * DH_;

  // Q fragments (DH=128 -> 4 chunks of K=32), loaded once, head-major rows.
  v16bf qf[4];
  {
    const __bf16* qrow = Qb + ((size_t)bh * S_ + q0 + lm) * DH_;
    for (int kc = 0; kc < 4; ++kc) {
      union { v16bf v; v8bf h2[2]; } u;
      u.h2[0] = *(const v8bf*)(qrow + kc * 32 + lk * 8);
      u.h2[1] = *(const v8bf*)(qrow + kc * 32 + lk * 8 + 16);
      qf[kc] = u.v;
    }
  }

  float rm[8], rsum[8], dv[8];
  for (int e = 0; e < 8; ++e) { rm[e] = -1e30f; rsum[e] = 0.f; dv[e] = 0.f; }

  if (wid == 0) {
    tdm_stage_k64x128(kbase, (unsigned)(size_t)&Ks[0][0]);
    tensor_wait0();
  }
  __syncthreads();

  int cur = 0;
  for (int kt0 = 0; kt0 < S_; kt0 += 64) {
    if (kt0 + 64 < S_ && wid == 0)   // TDM prefetch of next K block
      tdm_stage_k64x128(kbase + (size_t)(kt0 + 64) * DH_,
                        (unsigned)(size_t)&Ks[cur ^ 1][0]);

    for (int sub = 0; sub < 4; ++sub) {
      v8f acc = (v8f){0.f, 0.f, 0.f, 0.f, 0.f, 0.f, 0.f, 0.f};
      for (int kc = 0; kc < 4; ++kc) {
        union { v16bf v; v8bf h2[2]; } u;
        int row = sub * 16 + lm;
        u.h2[0] = *(const v8bf*)&Ks[cur][row * 136 + kc * 32 + lk * 8];
        u.h2[1] = *(const v8bf*)&Ks[cur][row * 136 + kc * 32 + lk * 8 + 16];
        acc = wmma_bf16(qf[kc], u.v, acc);
      }
      int colbase = kt0 + sub * 16 + lm;
      for (int e = 0; e < 8; ++e) {
        float sc = acc[e] * scale;
        float nm = fmaxf(rm[e], sc);
        rsum[e] = rsum[e] * __expf(rm[e] - nm) + __expf(sc - nm);
        rm[e] = nm;
        if (colbase == q0 + e + 8 * lk) dv[e] = sc;  // diagonal element
      }
    }
    if (wid == 0) tensor_wait0();   // prefetch DMA complete
    __syncthreads();                // all reads of cur buffer done (dscnt 0)
    cur ^= 1;
  }

  // Butterfly-merge the 16 lanes of each half-wave (rows stay within halves).
  for (int off = 1; off < 16; off <<= 1) {
    for (int e = 0; e < 8; ++e) {
      float om = __shfl_xor(rm[e], off, 32);
      float os = __shfl_xor(rsum[e], off, 32);
      float od = __shfl_xor(dv[e], off, 32);
      float nm = fmaxf(rm[e], om);
      rsum[e] = rsum[e] * __expf(rm[e] - nm) + os * __expf(om - nm);
      rm[e] = nm;
      dv[e] += od;  // exactly one lane held the diagonal, rest are 0
    }
  }
  if (lm == 0)
    for (int e = 0; e < 8; ++e)
      wsm[wid * 16 + e + 8 * lk] = __expf(dv[e] - rm[e]) / rsum[e];  // exp(diag-lse)
  __syncthreads();

  // out = w * v, packed bf16 in (B*S, D) for the O-projection GEMM.
  const size_t vbase = ((size_t)b * S_ + q0) * D_ + h * DH_;
  for (int idx = lane; idx < 16 * DH_; idx += 32) {
    int r = idx >> 7, c = idx & 127;
    float wv = wsm[wid * 16 + r];
    size_t off = vbase + (size_t)r * D_ + c;
    attnb[off] = f2bf(wv * Vf[off]);
  }
}

// ---------------- host side ----------------
extern "C" void kernel_launch(void* const* d_in, const int* in_sizes, int n_in,
                              void* d_out, int out_size, void* d_ws, size_t ws_size,
                              hipStream_t stream) {
  (void)in_sizes; (void)n_in; (void)out_size; (void)ws_size;
  const float* x   = (const float*)d_in[0];
  const float* rc  = (const float*)d_in[1];
  const float* rsn = (const float*)d_in[2];
  const float* Wq  = (const float*)d_in[3];
  const float* Wk  = (const float*)d_in[4];
  const float* Wv  = (const float*)d_in[5];
  const float* Wo  = (const float*)d_in[6];
  float* out = (float*)d_out;

  size_t off = 0;
  char* base = (char*)d_ws;
  auto alloc = [&](size_t bytes) -> void* {
    void* p = base + off;
    off += (bytes + 255) & ~(size_t)255;
    return p;
  };
  __bf16* xb    = (__bf16*)alloc((size_t)M_ * D_ * 2);
  __bf16* Wqb   = (__bf16*)alloc((size_t)D_ * D_ * 2);
  __bf16* Wkb   = (__bf16*)alloc((size_t)D_ * D_ * 2);
  __bf16* Wvb   = (__bf16*)alloc((size_t)D_ * D_ * 2);
  __bf16* Wob   = (__bf16*)alloc((size_t)D_ * D_ * 2);
  float*  Qf    = (float*)alloc((size_t)M_ * D_ * 4);
  float*  Kf    = (float*)alloc((size_t)M_ * D_ * 4);
  float*  Vf    = (float*)alloc((size_t)M_ * D_ * 4);
  __bf16* Qbh   = (__bf16*)alloc((size_t)M_ * D_ * 2);
  __bf16* Kbh   = (__bf16*)alloc((size_t)M_ * D_ * 2);
  __bf16* attnb = (__bf16*)alloc((size_t)M_ * D_ * 2);

  int nX = M_ * D_;
  int nW = D_ * D_;
  k_cvt<<<nX / 4 / 256, 256, 0, stream>>>(x, xb, nX);
  k_cvt<<<nW / 4 / 256, 256, 0, stream>>>(Wq, Wqb, nW);
  k_cvt<<<nW / 4 / 256, 256, 0, stream>>>(Wk, Wkb, nW);
  k_cvt<<<nW / 4 / 256, 256, 0, stream>>>(Wv, Wvb, nW);
  k_cvt<<<nW / 4 / 256, 256, 0, stream>>>(Wo, Wob, nW);

  dim3 gg(D_ / 128, M_ / 128);
  k_gemm_bt<<<gg, 256, 0, stream>>>(xb, Wqb, Qf, M_, D_, D_);
  k_gemm_bt<<<gg, 256, 0, stream>>>(xb, Wkb, Kf, M_, D_, D_);
  k_gemm_bt<<<gg, 256, 0, stream>>>(xb, Wvb, Vf, M_, D_, D_);

  int nR = B_ * S_ * H_ * (DH_ / 2);  // 4,194,304 rotation pairs
  k_rope<<<nR / 256, 256, 0, stream>>>(Qf, Kf, rc, rsn, Qbh, Kbh);

  dim3 ga(S_ / 128, B_ * H_);
  k_attn<<<ga, 256, 0, stream>>>(Qbh, Kbh, Vf, attnb);

  k_gemm_bt<<<gg, 256, 0, stream>>>(attnb, Wob, out, M_, D_, D_);
}
